// BidirectionalAttentionalPromptEncoder_1382979469928
// MI455X (gfx1250) — matrix operand order repaired and checked by reading, not compiled, not run on hardware
//
#include <hip/hip_runtime.h>
#include <math.h>

typedef __attribute__((ext_vector_type(16))) _Float16 v16h;
typedef __attribute__((ext_vector_type(8)))  float    v8f;
typedef __attribute__((ext_vector_type(4)))  unsigned int v4u;
typedef __attribute__((ext_vector_type(8)))  int      v8i;
typedef __attribute__((ext_vector_type(4)))  int      v4i;

#if defined(__has_builtin)
#if __has_builtin(__builtin_amdgcn_tensor_load_to_lds) && \
    __has_builtin(__builtin_amdgcn_s_wait_tensorcnt)
#define USE_TDM 1
#endif
#endif
#ifndef USE_TDM
#define USE_TDM 0
#endif

namespace {

constexpr int kC  = 1024;   // hidden size
constexpr int kT  = 64;     // seq len (P == S == 64)
constexpr int kB  = 64;     // batch
constexpr int kBT = kB * kT;      // 4096 rows
constexpr int kG  = 2048;         // 4*H2 gate width
constexpr int kH2 = 512;
constexpr int kPanelK = 128;      // GEMM k-panel staged in LDS via TDM

__device__ __forceinline__ float sigf(float x) { return 1.0f / (1.0f + expf(-x)); }

__device__ __forceinline__ v8f wmma16(v16h a, v16h b, v8f c) {
  return __builtin_amdgcn_wmma_f32_16x16x32_f16(false, a, false, b, (short)0, c,
                                                false, false);
}

// ---- WMMA fragment loaders (contiguous k = 16*(lane/16) + i in element i, wave32) ----
// A fragment: A row-major [M,K]; 16x32 tile at (m0,k0).
__device__ __forceinline__ v16h load_a(const _Float16* A, int lda, int m0, int k0) {
  int lane = threadIdx.x & 31;
  const _Float16* p = A + (size_t)(m0 + (lane & 15)) * lda + (k0 + ((lane >> 4) << 4));
  v16h r;
#pragma unroll
  for (int i = 0; i < 8; ++i) r[i] = p[i];
#pragma unroll
  for (int i = 0; i < 8; ++i) r[8 + i] = p[8 + i];
  return r;
}

// B fragment from W [N,K] row-major (computes A @ W^T). 32x16 tile (k0..k0+31, n0..n0+15).
__device__ __forceinline__ v16h load_bt(const _Float16* W, int ldw, int n0, int k0) {
  int lane = threadIdx.x & 31;
  const _Float16* p = W + (size_t)(n0 + (lane & 15)) * ldw + (k0 + ((lane >> 4) << 4));
  v16h r;
#pragma unroll
  for (int i = 0; i < 16; ++i) r[i] = p[i];
  return r;
}

// B fragment from B [K,N] row-major (non-transposed B, used for P@V).
__device__ __forceinline__ v16h load_bn(const _Float16* Bm, int ldb, int n0, int k0) {
  int lane = threadIdx.x & 31;
  const _Float16* p = Bm + (size_t)(k0 + ((lane >> 4) << 4)) * ldb + (n0 + (lane & 15));
  v16h r;
#pragma unroll
  for (int i = 0; i < 16; ++i) r[i] = p[(size_t)i * ldb];
  return r;
}

// ---- Tensor Data Mover: stage a [rows x cols] f16 tile (row stride ldw elems) into LDS ----
// D# per ISA ch.8: group0 {count=1, lds_addr, global_addr, type=2}; group1 {data_size=2B,
// tensor_dim0=cols, tensor_dim1=rows, tile_dim0=cols, tile_dim1=rows, dim0_stride=ldw}.
__device__ __forceinline__ void tdm_stage_b(const _Float16* gsrc, int ldw,
                                            unsigned ldsAddr, int rows, int cols) {
#if USE_TDM
  unsigned long long ga = (unsigned long long)(size_t)gsrc;
  v4u g0;
  g0[0] = 1u;                                            // count=1, user mode
  g0[1] = ldsAddr;                                       // lds_addr [63:32]
  g0[2] = (unsigned)(ga & 0xffffffffull);                // global_addr [95:64]
  g0[3] = (unsigned)((ga >> 32) & 0x01ffffffull) | (2u << 30);  // addr hi + type=2
  unsigned ur = (unsigned)rows, uc = (unsigned)cols;
  v8i g1;
  g1[0] = (int)0x00010000u;                              // data_size=1 (2 bytes)
  g1[1] = (int)((uc & 0xffffu) << 16);                   // tensor_dim0 lo16
  g1[2] = (int)(((uc >> 16) & 0xffffu) | ((ur & 0xffffu) << 16));  // dim0 hi | dim1 lo
  g1[3] = (int)(((ur >> 16) & 0xffffu) | ((uc & 0xffffu) << 16));  // dim1 hi | tile_dim0
  g1[4] = (int)(ur & 0xffffu);                           // tile_dim1 (tile_dim2=0)
  g1[5] = (int)(unsigned)ldw;                            // tensor_dim0_stride lo32
  g1[6] = 0;
  g1[7] = 0;
  v4i gz = {};
#if __clang_major__ >= 23
  v8i gz8 = {};
  __builtin_amdgcn_tensor_load_to_lds(g0, g1, gz, gz, gz8, 0);
#else
  __builtin_amdgcn_tensor_load_to_lds(g0, g1, gz, gz, 0);
#endif
#else
  (void)gsrc; (void)ldw; (void)ldsAddr; (void)rows; (void)cols;
#endif
}

// Fallback synchronous cooperative stage (all 128 threads), 64 x kPanelK f16.
__device__ __forceinline__ void sync_stage_b(const _Float16* W, int ldw, _Float16* dst,
                                             int nb, int k0) {
  for (int i = threadIdx.x; i < 64 * (kPanelK / 4); i += 128) {
    int row = i / (kPanelK / 4);
    int c4  = (i % (kPanelK / 4)) << 2;
    const unsigned long long* s =
        (const unsigned long long*)(W + (size_t)(nb + row) * ldw + k0 + c4);
    *((unsigned long long*)(dst + row * kPanelK + c4)) = *s;
  }
}

// ---------------- generic GEMM: C[M,N] = A[M,K] @ W[N,K]^T (+bias0+bias1) ----------------
// grid = (N/64, M/64), block = 128 (4 waves). B k-panels staged in LDS (TDM, double buffered);
// wave w -> m-tile w, all 4 n-tiles read B frags from LDS.
__global__ __launch_bounds__(128) void gemm_wt(
    const _Float16* __restrict__ A, int lda,
    const _Float16* __restrict__ W, int ldw,
    const float* __restrict__ bias0, const float* __restrict__ bias1,
    float* __restrict__ Cf, _Float16* __restrict__ Ch, int ldc, int K) {
  __shared__ _Float16 bstage[2][64 * kPanelK];   // 2 x 16 KiB
  int lane = threadIdx.x & 31;
  int wid  = threadIdx.x >> 5;
  int m0 = blockIdx.y * 64 + wid * 16;
  int nb = blockIdx.x * 64;
  v8f acc[4] = {};
  const int P = K / kPanelK;

#if USE_TDM
  if (wid == 0)
    tdm_stage_b(W + (size_t)nb * ldw, ldw, (unsigned)(size_t)&bstage[0][0], 64, kPanelK);
#else
  sync_stage_b(W, ldw, &bstage[0][0], nb, 0);
#endif

  for (int p = 0; p < P; ++p) {
    if (p + 1 < P) {
#if USE_TDM
      if (wid == 0)
        tdm_stage_b(W + (size_t)nb * ldw + (p + 1) * kPanelK, ldw,
                    (unsigned)(size_t)&bstage[(p + 1) & 1][0], 64, kPanelK);
#else
      sync_stage_b(W, ldw, &bstage[(p + 1) & 1][0], nb, (p + 1) * kPanelK);
#endif
#if USE_TDM
      if (wid == 0) __builtin_amdgcn_s_wait_tensorcnt(1);  // panel p complete
#endif
    } else {
#if USE_TDM
      if (wid == 0) __builtin_amdgcn_s_wait_tensorcnt(0);
#endif
    }
    __syncthreads();  // staged panel visible to all waves
    const _Float16* bl = &bstage[p & 1][0];
#pragma unroll
    for (int kk = 0; kk < kPanelK; kk += 32) {
      v16h a = load_a(A, lda, m0, p * kPanelK + kk);
#pragma unroll
      for (int t = 0; t < 4; ++t) {
        v16h b = load_bt(bl, kPanelK, t * 16, kk);   // LDS read
        acc[t] = wmma16(a, b, acc[t]);
      }
    }
    __syncthreads();  // reads done before this buffer is re-staged
  }

  int rlo = (lane >> 4) * 8;
  int cil = lane & 15;
#pragma unroll
  for (int t = 0; t < 4; ++t) {
    int col = nb + t * 16 + cil;
    float bv = (bias0 ? bias0[col] : 0.0f) + (bias1 ? bias1[col] : 0.0f);
#pragma unroll
    for (int e = 0; e < 8; ++e) {
      int row = m0 + rlo + e;
      float v = acc[t][e] + bv;
      if (Cf) Cf[(size_t)row * ldc + col] = v;
      if (Ch) Ch[(size_t)row * ldc + col] = (_Float16)v;
    }
  }
}

// ---------------- persistent LSTM scan (one block per directional scan) ----------------
// grid = 8 scans: s -> seqvar=s>>1 (0=p,1=s,2=pflip,3=sflip), dir=s&1, stack=seqvar>>1.
// h kept in LDS (f16 64x512), c in WMMA-accumulator-layout registers.
__global__ __launch_bounds__(512) void lstm_scan(
    const _Float16* __restrict__ xwbase,   // 8 x [B*T, 2048] f16 (x@Wih^T + biases)
    const _Float16* __restrict__ whhbase,  // [2][2][2][2048,512] f16
    _Float16* __restrict__ outbase,        // 4 x [B,T,1024] f16 (fwd cols 0..511, bwd 512..1023)
    int layer) {
  __shared__ _Float16 hbuf[64 * 512];
  int s = blockIdx.x;
  int seqvar = s >> 1, dir = s & 1, stack = seqvar >> 1;
  const _Float16* xw  = xwbase + (size_t)s * kBT * kG;
  const _Float16* whh = whhbase + (size_t)((stack * 2 + layer) * 2 + dir) * kG * kH2;
  _Float16* out = outbase + (size_t)seqvar * kBT * kC + dir * kH2;

  int lane = threadIdx.x & 31, wid = threadIdx.x >> 5;
  int rlo = (lane >> 4) * 8, cil = lane & 15;
  for (int i = threadIdx.x; i < 64 * 512; i += 512) hbuf[i] = (_Float16)0.0f;
  float cacc[8][8];
  float hreg[8][8];
#pragma unroll
  for (int g = 0; g < 8; ++g)
#pragma unroll
    for (int e = 0; e < 8; ++e) cacc[g][e] = 0.0f;

  for (int step = 0; step < 64; ++step) {
    int time = dir ? (63 - step) : step;
    __syncthreads();  // hbuf (t-1) visible to all waves
    if (step + 1 < 64) {  // prefetch next timestep's gate pre-activations into cache
      int tn = dir ? (62 - step) : (step + 1);
      const _Float16* pf =
          xw + ((size_t)(threadIdx.x & 63) * kT + tn) * kG + ((threadIdx.x >> 6) & 7) * 256;
      __builtin_prefetch(pf, 0, 1);
    }
#pragma unroll
    for (int g = 0; g < 8; ++g) {
      int gg = wid * 8 + g;
      int m0 = (gg & 3) * 16;        // batch tile (rows 0..63)
      int j0 = (gg >> 2) * 16;       // hidden-unit tile (cols 0..511)
      v8f acc[4];
#pragma unroll
      for (int q = 0; q < 4; ++q) {  // init with x@Wih^T (+biases) for gates i,f,g,o
#pragma unroll
        for (int e = 0; e < 8; ++e) {
          int bm = m0 + rlo + e;     // batch index
          acc[q][e] = (float)xw[((size_t)bm * kT + time) * kG + q * kH2 + j0 + cil];
        }
      }
      for (int kk = 0; kk < kH2; kk += 32) {
        v16h a = load_a(hbuf, 512, m0, kk);
#pragma unroll
        for (int q = 0; q < 4; ++q) {
          v16h b = load_bt(whh, kH2, q * kH2 + j0, kk);
          acc[q] = wmma16(a, b, acc[q]);
        }
      }
#pragma unroll
      for (int e = 0; e < 8; ++e) {
        float zi = acc[0][e], zf = acc[1][e], zg = acc[2][e], zo = acc[3][e];
        float c = cacc[g][e];
        c = sigf(zf) * c + sigf(zi) * tanhf(zg);
        cacc[g][e] = c;
        hreg[g][e] = sigf(zo) * tanhf(c);
      }
    }
    __syncthreads();  // all reads of old h done before overwrite
#pragma unroll
    for (int g = 0; g < 8; ++g) {
      int gg = wid * 8 + g;
      int m0 = (gg & 3) * 16, j0 = (gg >> 2) * 16;
#pragma unroll
      for (int e = 0; e < 8; ++e) {
        int row = m0 + rlo + e, col = j0 + cil;
        _Float16 hv = (_Float16)hreg[g][e];
        hbuf[row * 512 + col] = hv;
        out[((size_t)row * kT + time) * kC + col] = hv;
      }
    }
  }
}

// ---------------- attention core: per (batch, head) block ----------------
// grid = B*8, block = 128 (4 waves). S=QK^T/sqrt(128) -> softmax(rows) -> ctx = P@V.
__global__ __launch_bounds__(128) void attn_core(
    const _Float16* __restrict__ Q, const _Float16* __restrict__ Kx,
    const _Float16* __restrict__ V, _Float16* __restrict__ ctx) {
  __shared__ float S[64][68];
  __shared__ _Float16 P[64][64];
  int b = blockIdx.x >> 3, h = blockIdx.x & 7;
  int lane = threadIdx.x & 31, wid = threadIdx.x >> 5;
  int rlo = (lane >> 4) * 8, cil = lane & 15;
  const size_t base = (size_t)b * 64 * kC + h * 128;
  const _Float16* Qb = Q + base;
  const _Float16* Kb = Kx + base;
  const _Float16* Vb = V + base;
  {
    int m0 = wid * 16;
#pragma unroll
    for (int nt = 0; nt < 4; ++nt) {
      v8f acc = {};
#pragma unroll
      for (int k0 = 0; k0 < 128; k0 += 32) {
        v16h a = load_a(Qb, kC, m0, k0);
        v16h bb = load_bt(Kb, kC, nt * 16, k0);
        acc = wmma16(a, bb, acc);
      }
#pragma unroll
      for (int e = 0; e < 8; ++e)
        S[m0 + rlo + e][nt * 16 + cil] = acc[e] * 0.08838834764831845f;  // 1/sqrt(128)
    }
  }
  __syncthreads();
  if (threadIdx.x < 64) {
    int r = threadIdx.x;
    float mx = -1e30f;
    for (int j = 0; j < 64; ++j) mx = fmaxf(mx, S[r][j]);
    float sum = 0.0f;
    for (int j = 0; j < 64; ++j) sum += expf(S[r][j] - mx);
    float inv = 1.0f / sum;
    for (int j = 0; j < 64; ++j) P[r][j] = (_Float16)(expf(S[r][j] - mx) * inv);
  }
  __syncthreads();
  {
    int m0 = wid * 16;
#pragma unroll
    for (int nt = 0; nt < 8; ++nt) {
      v8f acc = {};
#pragma unroll
      for (int k0 = 0; k0 < 64; k0 += 32) {
        v16h a = load_a(&P[0][0], 64, m0, k0);
        v16h bb = load_bn(Vb, kC, nt * 16, k0);
        acc = wmma16(a, bb, acc);
      }
#pragma unroll
      for (int e = 0; e < 8; ++e)
        ctx[base + (size_t)(m0 + rlo + e) * kC + nt * 16 + cil] = (_Float16)acc[e];
    }
  }
}

// ---------------- layernorm (+ optional residual, + optional exact-erf GELU) ----------------
__global__ __launch_bounds__(256) void ln_kernel(
    const float* __restrict__ X, const float* __restrict__ Res,
    const float* __restrict__ gamma, const float* __restrict__ beta,
    float* __restrict__ Of, _Float16* __restrict__ Oh, int width, int doGelu) {
  int row = blockIdx.x * 8 + (threadIdx.x >> 5);
  int lane = threadIdx.x & 31;
  const float* x = X + (size_t)row * width;
  const float* r = Res ? Res + (size_t)row * width : nullptr;
  float s = 0.0f, s2 = 0.0f;
  for (int c = lane; c < width; c += 32) {
    float v = x[c] + (r ? r[c] : 0.0f);
    s += v; s2 += v * v;
  }
#pragma unroll
  for (int o = 16; o > 0; o >>= 1) {
    s  += __shfl_xor(s, o, 32);
    s2 += __shfl_xor(s2, o, 32);
  }
  float mean = s / width;
  float var  = s2 / width - mean * mean;
  float inv  = rsqrtf(var + 1e-5f);
  for (int c = lane; c < width; c += 32) {
    float v = x[c] + (r ? r[c] : 0.0f);
    v = (v - mean) * inv * gamma[c] + beta[c];
    if (doGelu) v = 0.5f * v * (1.0f + erff(v * 0.7071067811865475f));
    if (Of) Of[(size_t)row * width + c] = v;
    if (Oh) Oh[(size_t)row * width + c] = (_Float16)v;
  }
}

// ---------------- small elementwise kernels ----------------
__global__ void f32_to_f16_kernel(const float* __restrict__ s, _Float16* __restrict__ d, int n) {
  int i = blockIdx.x * blockDim.x + threadIdx.x;
  if (i < n) d[i] = (_Float16)s[i];
}

// emb + sinusoid -> f16; writes both sequence variants and their time flips, broadcast over batch
__global__ void build_inputs(const float* __restrict__ pemb, const float* __restrict__ semb,
                             _Float16* __restrict__ x16base) {
  int idx = blockIdx.x * blockDim.x + threadIdx.x;  // 2 * 64 * 1024
  if (idx >= 2 * kT * kC) return;
  int seq = idx >> 16;
  int t = (idx >> 10) & 63;
  int c = idx & 1023;
  const float* emb = seq ? semb : pemb;
  float i2 = (float)((c >> 1) << 1);
  float dv = expf(-logf(10000.0f) * i2 / (float)kC);
  float arg = (float)t * dv;
  float pe = (c & 1) ? cosf(arg) : sinf(arg);
  _Float16 v = (_Float16)(emb[t * kC + c] + pe);
  _Float16* xs = x16base + (size_t)seq * kBT * kC;
  _Float16* xf = x16base + (size_t)(seq + 2) * kBT * kC;
  int tf = 63 - t;
  for (int b = 0; b < kB; ++b) {
    xs[((size_t)b * kT + t)  * kC + c] = v;
    xf[((size_t)b * kT + tf) * kC + c] = v;
  }
}

// cat16[r, 0:1024] = f[r]; cat16[r, 1024:2048] = b[time-flip(r)]
__global__ void gate_concat(const _Float16* __restrict__ f, const _Float16* __restrict__ bsrc,
                            _Float16* __restrict__ cat) {
  int idx = blockIdx.x * blockDim.x + threadIdx.x;  // kBT * 2048
  if (idx >= kBT * kG) return;
  int row = idx >> 11, c = idx & 2047;
  if (c < kC) {
    cat[idx] = f[(size_t)row * kC + c];
  } else {
    int bi = row >> 6, t = row & 63;
    int frow = (bi << 6) | (63 - t);
    cat[idx] = bsrc[(size_t)frow * kC + (c - kC)];
  }
}

// bi = sigmoid(gates[:, :C]) * f + sigmoid(gates[:, C:]) * b  (f,b read from cat16)
__global__ void gate_combine(const float* __restrict__ gates, const _Float16* __restrict__ cat,
                             float* __restrict__ biF, _Float16* __restrict__ biH) {
  int idx = blockIdx.x * blockDim.x + threadIdx.x;  // kBT * kC
  if (idx >= kBT * kC) return;
  int row = idx >> 10, c = idx & 1023;
  size_t ro = (size_t)row * kG;
  float fg = sigf(gates[ro + c]);
  float bg = sigf(gates[ro + kC + c]);
  float v = fg * (float)cat[ro + c] + bg * (float)cat[ro + kC + c];
  biF[idx] = v;
  biH[idx] = (_Float16)v;
}

}  // namespace

extern "C" void kernel_launch(void* const* d_in, const int* in_sizes, int n_in,
                              void* d_out, int out_size, void* d_ws, size_t ws_size,
                              hipStream_t stream) {
  (void)in_sizes; (void)n_in; (void)out_size; (void)ws_size;
  const float* prefix_emb = (const float*)d_in[0];
  const float* suffix_emb = (const float*)d_in[1];
  const float* lstm_Wih   = (const float*)d_in[2];
  const float* lstm_Whh   = (const float*)d_in[3];
  const float* lstm_bih   = (const float*)d_in[4];
  const float* lstm_bhh   = (const float*)d_in[5];
  const float* attn_w     = (const float*)d_in[6];
  const float* attn_b     = (const float*)d_in[7];
  const float* gate_w1    = (const float*)d_in[8];
  const float* gate_b1    = (const float*)d_in[9];
  const float* gate_lg    = (const float*)d_in[10];
  const float* gate_lb    = (const float*)d_in[11];
  const float* gate_w2    = (const float*)d_in[12];
  const float* gate_b2    = (const float*)d_in[13];
  const float* out_w1     = (const float*)d_in[14];
  const float* out_b1     = (const float*)d_in[15];
  const float* out_g1     = (const float*)d_in[16];
  const float* out_be1    = (const float*)d_in[17];
  const float* out_w2     = (const float*)d_in[18];
  const float* out_b2     = (const float*)d_in[19];
  const float* out_g2     = (const float*)d_in[20];
  const float* out_be2    = (const float*)d_in[21];
  const float* ln_g       = (const float*)d_in[22];
  const float* ln_b       = (const float*)d_in[23];
  float* outp = (float*)d_out;

  const int szWih = 2 * 2 * 2 * kG * kC;
  const int szWhh = 2 * 2 * 2 * kG * kH2;
  const int szAW  = 2 * 4 * kC * kC;
  const int szG1  = 2 * kC * kG;
  const int szG2  = 2 * kG * kC;
  const int szO1  = 2 * kG * kC;
  const int szO2  = 2 * kC * kG;

  char* ws = (char*)d_ws;
  size_t off = 0;
  auto alloc = [&](size_t bytes) -> char* {
    char* p = ws + off;
    off += (bytes + 255) & ~(size_t)255;
    return p;
  };

  _Float16* wih16 = (_Float16*)alloc((size_t)szWih * 2);
  _Float16* whh16 = (_Float16*)alloc((size_t)szWhh * 2);
  _Float16* aw16  = (_Float16*)alloc((size_t)szAW * 2);
  _Float16* gw116 = (_Float16*)alloc((size_t)szG1 * 2);
  _Float16* gw216 = (_Float16*)alloc((size_t)szG2 * 2);
  _Float16* ow116 = (_Float16*)alloc((size_t)szO1 * 2);
  _Float16* ow216 = (_Float16*)alloc((size_t)szO2 * 2);
  _Float16* x16      = (_Float16*)alloc((size_t)4 * kBT * kC * 2);  // layer-0 inputs
  _Float16* lin1     = (_Float16*)alloc((size_t)4 * kBT * kC * 2);  // layer-1 inputs
  _Float16* lstmout  = (_Float16*)alloc((size_t)4 * kBT * kC * 2);  // final bilstm outputs
  char*     xwreg    = alloc((size_t)8 * kBT * kG * 2);             // 128 MiB x@Wih^T
  _Float16* bi16     = (_Float16*)alloc((size_t)2 * kBT * kC * 2);  // pbi, sbi (f16)
  _Float16* ctx16    = (_Float16*)alloc((size_t)kBT * kC * 2);
  float*    lnres32  = (float*)   alloc((size_t)2 * kBT * kC * 4);  // po/so after residual LN
  _Float16* lnres16  = (_Float16*)alloc((size_t)2 * kBT * kC * 2);

  // Post-LSTM scratch aliased into the xW region (dead after the scans complete).
  const size_t MiB = 1024 * 1024;
  _Float16* cat16 = (_Float16*)(xwreg + 0);          // [4096,2048] f16 (16 MiB)
  float*    tmpA  = (float*)   (xwreg + 16 * MiB);   // [4096,2048] f32 (32 MiB)
  float*    biF32 = (float*)   (xwreg + 48 * MiB);   // 2 x [4096,1024] f32 (32 MiB)
  _Float16* tmp16 = (_Float16*)(xwreg + 80 * MiB);   // [4096,2048] f16 (16 MiB)
  _Float16* q16   = (_Float16*)(xwreg + 96 * MiB);   // 3 x [4096,1024] f16 (24 MiB)
  _Float16* k16   = q16 + (size_t)kBT * kC;
  _Float16* v16   = k16 + (size_t)kBT * kC;

  auto conv = [&](const float* s, _Float16* d, int n) {
    f32_to_f16_kernel<<<(n + 255) / 256, 256, 0, stream>>>(s, d, n);
  };
  conv(lstm_Wih, wih16, szWih);
  conv(lstm_Whh, whh16, szWhh);
  conv(attn_w,   aw16,  szAW);
  conv(gate_w1,  gw116, szG1);
  conv(gate_w2,  gw216, szG2);
  conv(out_w1,   ow116, szO1);
  conv(out_w2,   ow216, szO2);

  build_inputs<<<(2 * kT * kC + 255) / 256, 256, 0, stream>>>(prefix_emb, suffix_emb, x16);

  // ---- bidirectional 2-layer LSTMs: precompute x@Wih^T then persistent scans ----
  for (int layer = 0; layer < 2; ++layer) {
    const _Float16* in = (layer == 0) ? x16 : lin1;
    _Float16* outb = (layer == 0) ? lin1 : lstmout;
    for (int s = 0; s < 8; ++s) {
      int seqvar = s >> 1, dir = s & 1, stack = seqvar >> 1;
      int widx = (stack * 2 + layer) * 2 + dir;
      gemm_wt<<<dim3(kG / 64, kBT / 64), 128, 0, stream>>>(
          in + (size_t)seqvar * kBT * kC, kC,
          wih16 + (size_t)widx * kG * kC, kC,
          lstm_bih + (size_t)widx * kG, lstm_bhh + (size_t)widx * kG,
          nullptr, (_Float16*)xwreg + (size_t)s * kBT * kG, kG, kC);
    }
    lstm_scan<<<8, 512, 0, stream>>>((const _Float16*)xwreg, whh16, outb, layer);
  }

  // ---- gates: bi = sig(g_f)*f + sig(g_b)*b ----
  for (int g = 0; g < 2; ++g) {
    gate_concat<<<(kBT * kG + 255) / 256, 256, 0, stream>>>(
        lstmout + (size_t)g * kBT * kC, lstmout + (size_t)(2 + g) * kBT * kC, cat16);
    gemm_wt<<<dim3(kC / 64, kBT / 64), 128, 0, stream>>>(
        cat16, kG, gw116 + (size_t)g * kC * kG, kG,
        gate_b1 + g * kC, nullptr, tmpA, nullptr, kC, kG);
    ln_kernel<<<kBT / 8, 256, 0, stream>>>(tmpA, nullptr, gate_lg + g * kC, gate_lb + g * kC,
                                           nullptr, tmp16, kC, 1);
    gemm_wt<<<dim3(kG / 64, kBT / 64), 128, 0, stream>>>(
        tmp16, kC, gw216 + (size_t)g * kG * kC, kC,
        gate_b2 + g * kG, nullptr, tmpA, nullptr, kG, kC);
    gate_combine<<<(kBT * kC + 255) / 256, 256, 0, stream>>>(
        tmpA, cat16, biF32 + (size_t)g * kBT * kC, bi16 + (size_t)g * kBT * kC);
  }

  // ---- cross attention (a=0: q=pbi,kv=sbi ; a=1: q=sbi,kv=pbi) + residual LN ----
  for (int a = 0; a < 2; ++a) {
    const _Float16* qx = bi16 + (size_t)((a == 0) ? 0 : 1) * kBT * kC;
    const _Float16* kv = bi16 + (size_t)((a == 0) ? 1 : 0) * kBT * kC;
    gemm_wt<<<dim3(kC / 64, kBT / 64), 128, 0, stream>>>(
        qx, kC, aw16 + (size_t)(a * 4 + 0) * kC * kC, kC,
        attn_b + (a * 4 + 0) * kC, nullptr, nullptr, q16, kC, kC);
    gemm_wt<<<dim3(kC / 64, kBT / 64), 128, 0, stream>>>(
        kv, kC, aw16 + (size_t)(a * 4 + 1) * kC * kC, kC,
        attn_b + (a * 4 + 1) * kC, nullptr, nullptr, k16, kC, kC);
    gemm_wt<<<dim3(kC / 64, kBT / 64), 128, 0, stream>>>(
        kv, kC, aw16 + (size_t)(a * 4 + 2) * kC * kC, kC,
        attn_b + (a * 4 + 2) * kC, nullptr, nullptr, v16, kC, kC);
    attn_core<<<kB * 8, 128, 0, stream>>>(q16, k16, v16, ctx16);
    gemm_wt<<<dim3(kC / 64, kBT / 64), 128, 0, stream>>>(
        ctx16, kC, aw16 + (size_t)(a * 4 + 3) * kC * kC, kC,
        attn_b + (a * 4 + 3) * kC, nullptr, tmpA, nullptr, kC, kC);
    ln_kernel<<<kBT / 8, 256, 0, stream>>>(tmpA, biF32 + (size_t)a * kBT * kC, ln_g, ln_b,
                                           lnres32 + (size_t)a * kBT * kC,
                                           lnres16 + (size_t)a * kBT * kC, kC, 0);
  }

  // ---- output transforms -> d_out (po then so, f32) ----
  for (int sd = 0; sd < 2; ++sd) {
    gemm_wt<<<dim3(kG / 64, kBT / 64), 128, 0, stream>>>(
        lnres16 + (size_t)sd * kBT * kC, kC, ow116 + (size_t)sd * kG * kC, kC,
        out_b1 + sd * kG, nullptr, tmpA, nullptr, kG, kC);
    ln_kernel<<<kBT / 8, 256, 0, stream>>>(tmpA, nullptr, out_g1 + sd * kG, out_be1 + sd * kG,
                                           nullptr, tmp16, kG, 1);
    gemm_wt<<<dim3(kC / 64, kBT / 64), 128, 0, stream>>>(
        tmp16, kG, ow216 + (size_t)sd * kC * kG, kG,
        out_b2 + sd * kC, nullptr, tmpA, nullptr, kC, kG);
    ln_kernel<<<kBT / 8, 256, 0, stream>>>(tmpA, nullptr, out_g2 + sd * kC, out_be2 + sd * kC,
                                           outp + (size_t)sd * kBT * kC, nullptr, kC, 0);
  }
}
